// ConvLSTM_58325655879904
// MI455X (gfx1250) — matrix operand order repaired and checked by reading, not compiled, not run on hardware
//
#include <hip/hip_runtime.h>
#include <hip/hip_bf16.h>

// ---------------------------------------------------------------------------
// Sizes (fixed by the reference)
// ---------------------------------------------------------------------------
constexpr int B_   = 64;
constexpr int CIN  = 256;
constexpr int S_   = 2048;
constexpr int HC   = 512;
constexpr int KDIM = CIN + HC;     // 768
constexpr int G4   = 4 * HC;       // 2048 gate rows

typedef __bf16 bf16_t;
typedef __attribute__((ext_vector_type(16))) __bf16 v16bf;
typedef __attribute__((ext_vector_type(8)))  float  v8f;
typedef __attribute__((ext_vector_type(4)))  float  f4_t;

union BF16x16 {           // 32 bytes: one WMMA bf16 A/B fragment per lane
  v16bf v;
  f4_t  f4[2];
};

// Workspace layout (bytes)
constexpr size_t WEFF_BYTES = (size_t)G4 * KDIM * 2;          // 3,145,728
constexpr size_t XT_BYTES   = (size_t)S_ * B_ * CIN * 2;      // 67,108,864
constexpr size_t HBF_BYTES  = (size_t)B_ * HC * 2;            // 65,536
constexpr size_t WEFF_OFF = 0;
constexpr size_t XT_OFF   = WEFF_OFF + WEFF_BYTES;
constexpr size_t HBF_OFF  = XT_OFF + XT_BYTES;
constexpr size_t BAR_OFF  = HBF_OFF + HBF_BYTES;

__device__ __forceinline__ unsigned short f32_to_bf16(float f) {
  unsigned u = __builtin_bit_cast(unsigned, f);
  unsigned r = u + 0x7FFFu + ((u >> 16) & 1u);   // round-to-nearest-even
  return (unsigned short)(r >> 16);
}
__device__ __forceinline__ float fast_sigmoid(float x) {
  return 1.0f / (1.0f + __expf(-x));
}
__device__ __forceinline__ float fast_tanh(float x) {
  float ax = fabsf(x);
  float e  = __expf(2.0f * ax);
  float t  = 1.0f - 2.0f / (e + 1.0f);           // e->inf => t->1 (no NaN)
  return copysignf(t, x);
}

// ---------------------------------------------------------------------------
// Kernel 0: zero h-state and barrier words (workspace is poisoned by harness)
// ---------------------------------------------------------------------------
__global__ void init_state(unsigned* __restrict__ h32, unsigned* __restrict__ bar) {
  int tid = blockIdx.x * blockDim.x + threadIdx.x;
  for (int i = tid; i < (int)(HBF_BYTES / 4); i += blockDim.x * gridDim.x) h32[i] = 0u;
  if (tid < 8) bar[tid] = 0u;
}

// ---------------------------------------------------------------------------
// Kernel 1: W_eff[g][c] = bf16(W[g][c][1])   (only center tap matters)
// ---------------------------------------------------------------------------
__global__ void prep_weights(const float* __restrict__ W, unsigned short* __restrict__ Weff) {
  int gid = blockIdx.x * blockDim.x + threadIdx.x;     // over 2048*768
  if (gid < G4 * KDIM) {
    Weff[gid] = f32_to_bf16(W[(size_t)gid * 3 + 1]);
  }
}

// ---------------------------------------------------------------------------
// Kernel 2: transpose x [B,CIN,S] -> xT bf16 [S][B][CIN] via LDS 32x32 tiles
// ---------------------------------------------------------------------------
__global__ void prep_x(const float* __restrict__ x, unsigned short* __restrict__ xT) {
  __shared__ float tile[32][33];
  const int t0 = blockIdx.x * 32;
  const int c0 = blockIdx.y * 32;
  const int b  = blockIdx.z;
  const int tx = threadIdx.x, ty = threadIdx.y;
  tile[ty][tx] = x[((size_t)b * CIN + (c0 + ty)) * S_ + (t0 + tx)];
  __syncthreads();
  xT[((size_t)(t0 + ty) * B_ + b) * CIN + (c0 + tx)] = f32_to_bf16(tile[tx][ty]);
}

// ---------------------------------------------------------------------------
// One K-slab: A fragment (16x32 bf16) x 4 gate B fragments -> 4 accumulators
// ---------------------------------------------------------------------------
__device__ __forceinline__ void wmma_kstep(const bf16_t* __restrict__ ap,
                                           const bf16_t* __restrict__ wp,
                                           v8f& ai, v8f& af, v8f& ao, v8f& ag) {
  constexpr size_t GS = (size_t)HC * KDIM;  // gate stride in W_eff rows*K
  BF16x16 a, b;
  a.f4[0] = *reinterpret_cast<const f4_t*>(ap);        // K = k0 + 16*(lane/16) .. +7
  a.f4[1] = *reinterpret_cast<const f4_t*>(ap + 8);    // K = k0 + 16*(lane/16) + 8 .. +15
  b.f4[0] = *reinterpret_cast<const f4_t*>(wp);
  b.f4[1] = *reinterpret_cast<const f4_t*>(wp + 8);
  ai = __builtin_amdgcn_wmma_f32_16x16x32_bf16(false, a.v, false, b.v, (short)0, ai, false, false);
  b.f4[0] = *reinterpret_cast<const f4_t*>(wp + GS);
  b.f4[1] = *reinterpret_cast<const f4_t*>(wp + GS + 8);
  af = __builtin_amdgcn_wmma_f32_16x16x32_bf16(false, a.v, false, b.v, (short)0, af, false, false);
  b.f4[0] = *reinterpret_cast<const f4_t*>(wp + 2 * GS);
  b.f4[1] = *reinterpret_cast<const f4_t*>(wp + 2 * GS + 8);
  ao = __builtin_amdgcn_wmma_f32_16x16x32_bf16(false, a.v, false, b.v, (short)0, ao, false, false);
  b.f4[0] = *reinterpret_cast<const f4_t*>(wp + 3 * GS);
  b.f4[1] = *reinterpret_cast<const f4_t*>(wp + 3 * GS + 8);
  ag = __builtin_amdgcn_wmma_f32_16x16x32_bf16(false, a.v, false, b.v, (short)0, ag, false, false);
}

// ---------------------------------------------------------------------------
// Kernel 3: persistent recurrent LSTM. 16 blocks x 8 waves = 128 waves;
// wave (m_tile, j_tile) owns rows b = m_tile*16..+15, cols j = j_tile*16..+15
// and all 4 gate tiles for them. c-state lives in VGPRs across all steps.
// ---------------------------------------------------------------------------
__global__ __launch_bounds__(256, 1) void lstm_seq(
    const bf16_t* __restrict__ Weff,   // [2048][768] bf16 row-major
    const bf16_t* __restrict__ xT,     // [S][64][256] bf16
    const float*  __restrict__ bias,   // [2048] f32
    bf16_t*       __restrict__ hbf,    // [64][512] bf16 (recurrent state)
    float*        __restrict__ out,    // [64][512][2048] f32
    unsigned*     __restrict__ bar)    // bar[0]=arrive count, bar[1]=generation
{
  const int lane   = threadIdx.x & 31;
  const int wave   = (blockIdx.x << 3) | (threadIdx.x >> 5);  // 0..127
  const int m_tile = wave & 3;
  const int j0     = (wave >> 2) << 4;        // 0..496
  const int lmod   = lane & 15;
  const int lhi    = lane >> 4;               // 0/1
  const int arow   = m_tile * 16 + lmod;      // A-fragment row (batch index)

  // bias depends only on the C-tile column N = lane%16 -> load once, keep in regs
  float bia_i = bias[0 * HC + j0 + lmod];
  float bia_f = bias[1 * HC + j0 + lmod];
  float bia_o = bias[2 * HC + j0 + lmod];
  float bia_g = bias[3 * HC + j0 + lmod];

  float cst[8];
  #pragma unroll
  for (int r = 0; r < 8; ++r) cst[r] = 0.0f;

  const bf16_t* hrow  = hbf + (size_t)arow * HC;
  const size_t  wbase = (size_t)(j0 + lmod) * KDIM + (size_t)lhi * 16;

  unsigned gen_target = 1;

  for (int t = 0; t < S_; ++t) {
    v8f ai = {}, af = {}, ao = {}, ag = {};
    const bf16_t* xrow = xT + ((size_t)t * B_ + arow) * CIN;

    // --- x contribution: K = 0..255 ---
    #pragma unroll 2
    for (int kk = 0; kk < CIN / 32; ++kk) {
      const int k0 = kk * 32;
      wmma_kstep(xrow + k0 + lhi * 16, Weff + wbase + k0, ai, af, ao, ag);
    }
    // --- h contribution: K = 256..767 ---
    #pragma unroll 2
    for (int kk = 0; kk < HC / 32; ++kk) {
      const int k0 = kk * 32;
      wmma_kstep(hrow + k0 + lhi * 16, Weff + wbase + CIN + k0, ai, af, ao, ag);
    }

    // --- elementwise LSTM cell update; C layout: (M = r + 8*lhi, N = lmod) ---
    #pragma unroll
    for (int r = 0; r < 8; ++r) {
      float i_ = fast_sigmoid(ai[r] + bia_i);
      float f_ = fast_sigmoid(af[r] + bia_f);
      float o_ = fast_sigmoid(ao[r] + bia_o);
      float g_ = fast_tanh   (ag[r] + bia_g);
      float cn = f_ * cst[r] + i_ * g_;
      cst[r]   = cn;
      float hn = o_ * fast_tanh(cn);
      const int bidx = m_tile * 16 + r + 8 * lhi;
      const int jidx = j0 + lmod;
      out[((size_t)bidx * HC + jidx) * S_ + t] = hn;
      unsigned short hb = f32_to_bf16(hn);
      *((unsigned short*)hbf + (size_t)bidx * HC + jidx) = hb;
    }

    // hint: pull next timestep's x slice toward the caches (global_prefetch_b8)
    if (t + 1 < S_) {
      __builtin_prefetch(xT + ((size_t)(t + 1) * B_ + arow) * CIN + lhi * 64, 0, 1);
    }

    // --- grid-wide barrier: h(t) must be visible before any wave starts t+1 ---
    __threadfence();
    __syncthreads();
    if (threadIdx.x == 0) {
      unsigned old = __hip_atomic_fetch_add(&bar[0], 1u, __ATOMIC_ACQ_REL,
                                            __HIP_MEMORY_SCOPE_AGENT);
      if (old == gridDim.x - 1) {
        __hip_atomic_store(&bar[0], 0u, __ATOMIC_RELAXED, __HIP_MEMORY_SCOPE_AGENT);
        __hip_atomic_fetch_add(&bar[1], 1u, __ATOMIC_RELEASE, __HIP_MEMORY_SCOPE_AGENT);
      } else {
        while (__hip_atomic_load(&bar[1], __ATOMIC_ACQUIRE,
                                 __HIP_MEMORY_SCOPE_AGENT) < gen_target) { }
      }
    }
    __syncthreads();
    ++gen_target;
  }
}

// ---------------------------------------------------------------------------
extern "C" void kernel_launch(void* const* d_in, const int* in_sizes, int n_in,
                              void* d_out, int out_size, void* d_ws, size_t ws_size,
                              hipStream_t stream) {
  const float* x    = (const float*)d_in[0];   // [64, 256, 2048]
  const float* W    = (const float*)d_in[1];   // [2048, 768, 3]
  const float* bias = (const float*)d_in[2];   // [2048]
  float*       out  = (float*)d_out;           // [64, 512, 2048]

  char* ws = (char*)d_ws;
  bf16_t*   Weff = (bf16_t*)(ws + WEFF_OFF);
  bf16_t*   xT   = (bf16_t*)(ws + XT_OFF);
  bf16_t*   hbf  = (bf16_t*)(ws + HBF_OFF);
  unsigned* bar  = (unsigned*)(ws + BAR_OFF);

  (void)in_sizes; (void)n_in; (void)out_size; (void)ws_size;

  init_state<<<dim3(64), dim3(256), 0, stream>>>((unsigned*)hbf, bar);

  const int wtot = G4 * KDIM;
  prep_weights<<<dim3((wtot + 255) / 256), dim3(256), 0, stream>>>(W, (unsigned short*)Weff);

  prep_x<<<dim3(S_ / 32, CIN / 32, B_), dim3(32, 32, 1), 0, stream>>>(x, (unsigned short*)xT);

  lstm_seq<<<dim3(16), dim3(256), 0, stream>>>(Weff, xT, bias, hbf, out, bar);
}
